// UltraJointModel_28595892257414
// MI455X (gfx1250) — compile-verified, matrix operand order checked by reading, not run on hardware
//
#include <hip/hip_runtime.h>
#include <hip/hip_bf16.h>
#include <math.h>

typedef __attribute__((ext_vector_type(16))) _Float16 v16h;
typedef __attribute__((ext_vector_type(8)))  float    v8f;
typedef __attribute__((ext_vector_type(8)))  unsigned u32x8;

#define HDIM  448
#define INDIM 1280
#define SEQL  203
#define PSEQ  217          // SEQL + 2*7 zero pad (max k=15)
#define NB    64
#define NNODE (NB*SEQL)    // 12992
#define NHEADS 16
#define HD    28
#define D3    1344
#define GATH  8
#define GHD   56

// ---------------- device helpers ----------------
__device__ __forceinline__ float gelu_f(float x){
  return 0.5f * x * (1.0f + erff(x * 0.70710678118654752f));
}
__device__ __forceinline__ float sigm(float x){ return 1.0f/(1.0f+expf(-x)); }
__device__ __forceinline__ unsigned fenc(float f){
  unsigned u = __float_as_uint(f);
  return (u & 0x80000000u) ? ~u : (u | 0x80000000u);
}
__device__ __forceinline__ float fdec(unsigned e){
  return (e & 0x80000000u) ? __uint_as_float(e & 0x7fffffffu) : __uint_as_float(~e);
}
// fragment-major permutation: lane<16 consumes K-pairs {0..3,8..11} (-> slots 0..7),
// lane>=16 consumes {4..7,12..15} (-> slots 8..15)
__device__ __forceinline__ int fragpos(int cp){
  return (cp & 4) ? ((cp & 8) ? cp : cp + 4)
                  : ((cp & 8) ? cp - 4 : cp);
}

// ---------------- WMMA GEMM ----------------
// C(MxN) = A(MxK,f32) * B(KxN, packed f16 pairs) [+ bias / gelu / residual / bn]
// epi: 0 raw, 1 +bias, 2 gelu(+bias), 3 +bias+residual, 4 conv: gelu((x+b)*rs*bng+bnb)
// accin: preload C tile into accumulators (tap accumulation for convs)
// aseq>0: A row remap  arow = (m/aseq)*apseq + m%aseq + aoff  (padded conv input)
__global__ __launch_bounds__(128)
void k_gemm(const float* __restrict__ A, const unsigned* __restrict__ Bp,
            const float* __restrict__ bias, const float* __restrict__ resid,
            const float* __restrict__ bng, const float* __restrict__ bnb,
            float* __restrict__ C,
            int M, int N, int K, int lda, int ldc, int ccol0,
            int accin, int epi, int aseq, int apseq, int aoff)
{
  // fragment-major tiles: row stride 20 dwords keeps 16B alignment for b128
  // loads and spreads the 16 lane base addresses across distinct LDS banks.
  __shared__ unsigned As[64][20];   // [row][frag slot 0..15, pad 4]
  __shared__ unsigned Bs[64][20];   // [col][frag slot 0..15, pad 4]
  const int tid  = threadIdx.x;
  const int lane = tid & 31;
  const int wv   = tid >> 5;
  const int tM   = blockIdx.y * 64;
  const int tN   = blockIdx.x * 64;
  const int mhi  = (lane < 16) ? 0 : 8;   // C/D row-half per lane group
  const int nloc = lane & 15;
  const int fb   = (lane < 16) ? 0 : 8;   // fragment slot base per lane group

  v8f zero = {0.f,0.f,0.f,0.f,0.f,0.f,0.f,0.f};
  v8f acc[4]; acc[0]=zero; acc[1]=zero; acc[2]=zero; acc[3]=zero;
  if (accin) {
    for (int s=0;s<4;s++)
      for (int i=0;i<8;i++){
        int gr = tM + wv*16 + mhi + i;
        int gc = tN + s*16 + nloc;
        if (gr < M && gc < N) acc[s][i] = C[(size_t)gr*ldc + ccol0 + gc];
      }
  }

  const int nkt = K >> 5;
  for (int kt=0; kt<nkt; ++kt){
    const int k0 = kt << 5;
    { // cooperative A tile load + f32->f16 pack (fragment-major store)
      int ibase = tid * 8;
      #pragma unroll
      for (int u=0; u<8; ++u){
        int i = ibase + u;
        int r = i >> 4, cp = i & 15;
        int m = tM + r;
        float ax = 0.f, ay = 0.f;
        if (m < M){
          int ar = m;
          if (aseq > 0){ int bb = m / aseq; ar = bb*apseq + (m - bb*aseq) + aoff; }
          const float* ap = A + (size_t)ar*lda + k0 + cp*2;
          ax = ap[0]; ay = ap[1];
          if (u == 0 && kt+1 < nkt) __builtin_prefetch(ap + 32, 0, 0);
        }
        union { _Float16 h[2]; unsigned u32; } pk;
        pk.h[0] = (_Float16)ax; pk.h[1] = (_Float16)ay;
        As[r][fragpos(cp)] = pk.u32;
      }
    }
    { // cooperative B tile load (pre-packed pairs, fragment-major store)
      int kp0 = k0 >> 1;
      int ibase = tid * 8;
      #pragma unroll
      for (int u=0; u<8; ++u){
        int i = ibase + u;
        int r = i >> 6, c = i & 63;
        int gc = tN + c;
        unsigned v = 0u;
        if (gc < N){
          v = Bp[(size_t)(kp0 + r)*N + gc];
          if (u == 0 && kt+1 < nkt) __builtin_prefetch(Bp + (size_t)(kp0 + 16 + r)*N + gc, 0, 0);
        }
        Bs[c][fragpos(r)] = v;
      }
    }
    __syncthreads();
    // A fragment: two ds_load_b128 (contiguous fragment-major slots)
    union { u32x8 v; v16h h; } af;
    af.v = *(const u32x8*)&As[wv*16 + nloc][fb];
    #pragma unroll
    for (int s=0;s<4;++s){
      union { u32x8 v; v16h h; } bf;
      bf.v = *(const u32x8*)&Bs[s*16 + nloc][fb];
      acc[s] = __builtin_amdgcn_wmma_f32_16x16x32_f16(false, af.h, false, bf.h,
                                                      (short)0, acc[s], false, false);
    }
    __syncthreads();
  }

  const float RSBN = 0.99999500003749969f;  // 1/sqrt(1+1e-5)
  for (int s=0;s<4;++s)
    for (int i=0;i<8;++i){
      int gr = tM + wv*16 + mhi + i;
      int gc = tN + s*16 + nloc;
      if (gr >= M || gc >= N) continue;
      float v = acc[s][i];
      if      (epi == 1) v += bias[gc];
      else if (epi == 2) v = gelu_f(v + bias[gc]);
      else if (epi == 3) v += bias[gc] + resid[(size_t)gr*ldc + ccol0 + gc];
      else if (epi == 4) v = gelu_f((v + bias[gc]) * RSBN * bng[gc] + bnb[gc]);
      C[(size_t)gr*ldc + ccol0 + gc] = v;
    }
}

// ---------------- weight packing ----------------
__global__ void k_pack_w(const float* __restrict__ W, unsigned* __restrict__ Bp,
                         int K, int n, int ldw, int col0)
{
  int idx = blockIdx.x*blockDim.x + threadIdx.x;
  int tot = (K>>1)*n;
  if (idx >= tot) return;
  int kp = idx / n, c = idx % n;
  float a = W[(size_t)(2*kp)  *ldw + col0 + c];
  float b = W[(size_t)(2*kp+1)*ldw + col0 + c];
  union { _Float16 h[2]; unsigned u; } p;
  p.h[0] = (_Float16)a; p.h[1] = (_Float16)b;
  Bp[idx] = p.u;
}

__global__ void k_pack_conv(const float* __restrict__ W, unsigned* __restrict__ Bp,
                            int ks, int tap)
{ // W: (64, 1280, ks)  ->  Bp: (640 pairs) x 64 out
  int idx = blockIdx.x*blockDim.x + threadIdx.x;
  if (idx >= (INDIM/2)*64) return;
  int kp = idx / 64, o = idx % 64;
  int i2 = kp*2;
  float a = W[((size_t)o*INDIM + i2  )*ks + tap];
  float b = W[((size_t)o*INDIM + i2+1)*ks + tap];
  union { _Float16 h[2]; unsigned u; } p;
  p.h[0] = (_Float16)a; p.h[1] = (_Float16)b;
  Bp[idx] = p.u;
}

// ---------------- layout / misc ----------------
__global__ void k_build_xp(const float* __restrict__ data, float* __restrict__ xp)
{ // data (B,1280,203) -> xp (B,217,1280) with 7-row zero pad (memset first)
  int idx = blockIdx.x*blockDim.x + threadIdx.x;
  if (idx >= NB*SEQL*INDIM) return;
  int c = idx % INDIM; int t = idx / INDIM;
  int l = t % SEQL; int b = t / SEQL;
  xp[((size_t)b*PSEQ + l + 7)*INDIM + c] = data[((size_t)b*INDIM + c)*SEQL + l];
}

__global__ void k_fill(float* p, float v, int n)
{ int i = blockIdx.x*blockDim.x + threadIdx.x; if (i<n) p[i]=v; }

__global__ void k_addip(float* a, const float* b, int n)
{ int i = blockIdx.x*blockDim.x + threadIdx.x; if (i<n) a[i]+=b[i]; }

__global__ void k_scale_rows(float* dst, const float* src, const float* s,
                             int L, int D, int total)
{ int i = blockIdx.x*blockDim.x + threadIdx.x; if (i>=total) return;
  int n = i / D; int b = n / L;
  dst[i] = src[i] * s[(size_t)b*D + (i % D)]; }

__global__ void k_colmean(const float* __restrict__ X, float* __restrict__ out,
                          int Bn, int L, int D, int ldo, int col0)
{ int i = blockIdx.x*blockDim.x + threadIdx.x; if (i>=Bn*D) return;
  int b = i / D, c = i % D; float s = 0.f;
  for (int l=0;l<L;++l) s += X[((size_t)b*L+l)*D + c];
  out[(size_t)b*ldo + col0 + c] = s/(float)L; }

__global__ void k_colmax(const float* __restrict__ X, float* __restrict__ out,
                         int Bn, int L, int D, int ldo, int col0)
{ int i = blockIdx.x*blockDim.x + threadIdx.x; if (i>=Bn*D) return;
  int b = i / D, c = i % D; float m = -3.0e38f;
  for (int l=0;l<L;++l) m = fmaxf(m, X[((size_t)b*L+l)*D + c]);
  out[(size_t)b*ldo + col0 + c] = m; }

// ---------------- layernorm (rows) ----------------
__global__ __launch_bounds__(256)
void k_ln(const float* __restrict__ X, const float* __restrict__ g,
          const float* __restrict__ b, float* __restrict__ Y, int D, int dogelu)
{
  int row = blockIdx.x;
  const float* x = X + (size_t)row * D;
  float* y = Y + (size_t)row * D;
  __shared__ float red[256];
  float s = 0.f;
  for (int i = threadIdx.x; i < D; i += 256) s += x[i];
  red[threadIdx.x] = s; __syncthreads();
  for (int o=128;o>0;o>>=1){ if (threadIdx.x<o) red[threadIdx.x]+=red[threadIdx.x+o]; __syncthreads(); }
  float mean = red[0] / (float)D; __syncthreads();
  float v = 0.f;
  for (int i = threadIdx.x; i < D; i += 256){ float d = x[i]-mean; v += d*d; }
  red[threadIdx.x] = v; __syncthreads();
  for (int o=128;o>0;o>>=1){ if (threadIdx.x<o) red[threadIdx.x]+=red[threadIdx.x+o]; __syncthreads(); }
  float rstd = rsqrtf(red[0]/(float)D + 1e-5f);
  for (int i = threadIdx.x; i < D; i += 256){
    float t = (x[i]-mean)*rstd*g[i] + b[i];
    y[i] = dogelu ? gelu_f(t) : t;
  }
}

// ---------------- small dense (SE etc.) ----------------
__global__ void k_small_nn(const float* __restrict__ A, const float* __restrict__ W,
                           const float* __restrict__ bias, float* __restrict__ C,
                           int M, int N, int K, int act)
{
  int idx = blockIdx.x*blockDim.x + threadIdx.x;
  if (idx >= M*N) return;
  int m = idx / N, n = idx % N;
  float s = bias ? bias[n] : 0.f;
  for (int k=0;k<K;++k) s += A[(size_t)m*K+k]*W[(size_t)k*N+n];
  if (act==1) s = gelu_f(s); else if (act==2) s = sigm(s);
  C[idx] = s;
}

// ---------------- GNN edge kernels ----------------
__global__ void k_edge_mlp(const float* __restrict__ eA, const float* __restrict__ eB,
                           const int* __restrict__ row, const int* __restrict__ col,
                           const float* __restrict__ w2, const float* __restrict__ b2,
                           float* __restrict__ ew, int E)
{
  int e = blockIdx.x*blockDim.x + threadIdx.x; if (e>=E) return;
  int r = row[e], cl = col[e];
  float s = b2[0];
  for (int j=0;j<64;++j)
    s += gelu_f(eA[(size_t)r*64+j] + eB[(size_t)cl*64+j]) * w2[j];
  ew[e] = sigm(s);
}

__global__ void k_deg_edge(float* deg, const int* col, const float* ew, int E)
{ int e = blockIdx.x*blockDim.x + threadIdx.x; if (e<E) atomicAdd(&deg[col[e]], ew[e]); }

__global__ void k_deg_fin(float* deg, int n)
{ int i = blockIdx.x*blockDim.x + threadIdx.x; if (i<n) deg[i] = (deg[i]>0.f) ? rsqrtf(deg[i]) : 0.f; }

__global__ void k_asad(const float* __restrict__ g, const float* __restrict__ as,
                       const float* __restrict__ ad, float* a_s, float* a_d, int Nn)
{
  int idx = blockIdx.x*blockDim.x + threadIdx.x; if (idx>=Nn*GATH) return;
  int n = idx / GATH, h = idx % GATH;
  float s=0.f, d=0.f;
  for (int j=0;j<GHD;++j){
    float gv = g[(size_t)n*HDIM + h*GHD + j];
    s += gv*as[h*GHD+j]; d += gv*ad[h*GHD+j];
  }
  a_s[idx]=s; a_d[idx]=d;
}

__global__ void k_gat_self(const float* a_s, const float* a_d, float* als,
                           unsigned* mxe, int Nn)
{ int i = blockIdx.x*blockDim.x + threadIdx.x; if (i>=Nn*GATH) return;
  float v = a_s[i]+a_d[i]; v = (v>0.f)? v : 0.2f*v;
  als[i]=v; mxe[i]=fenc(v); }

__global__ void k_gat_edge(const float* a_s, const float* a_d, const int* row,
                           const int* col, float* al, unsigned* mxe, int E)
{
  int e = blockIdx.x*blockDim.x + threadIdx.x; if (e>=E) return;
  int r = row[e], cl = col[e];
  for (int h=0;h<GATH;++h){
    float v = a_s[(size_t)r*GATH+h] + a_d[(size_t)cl*GATH+h];
    v = (v>0.f)? v : 0.2f*v;
    al[(size_t)e*GATH+h] = v;
    atomicMax(&mxe[(size_t)cl*GATH+h], fenc(v));
  }
}

__global__ void k_den_init(const float* als, const unsigned* mxe, float* den, int Nn)
{ int i = blockIdx.x*blockDim.x + threadIdx.x; if (i>=Nn*GATH) return;
  den[i] = expf(als[i] - fdec(mxe[i])); }

__global__ void k_den_edge(const float* al, const int* col, const unsigned* mxe,
                           float* den, int E)
{ int i = blockIdx.x*blockDim.x + threadIdx.x; if (i>=E*GATH) return;
  int e = i / GATH, h = i % GATH; int cl = col[e];
  atomicAdd(&den[(size_t)cl*GATH+h], expf(al[i] - fdec(mxe[(size_t)cl*GATH+h]))); }

__global__ void k_acc_init(const float* xw, const float* gg, const float* dis,
                           const float* als, const unsigned* mxe, const float* den,
                           const float* gcnb, const float* gatb, float* acc, int Nn)
{
  int idx = blockIdx.x*blockDim.x + threadIdx.x; if (idx>=Nn*HDIM) return;
  int n = idx / HDIM, c = idx % HDIM; int h = c / GHD;
  float attself = expf(als[(size_t)n*GATH+h] - fdec(mxe[(size_t)n*GATH+h]))
                  / (den[(size_t)n*GATH+h] + 1e-16f);
  float d = dis[n];
  acc[idx] = gcnb[c] + gatb[c] + xw[idx]*d*d + gg[idx]*attself;
}

__global__ void k_scatter(const float* __restrict__ xw, const float* __restrict__ gg,
                          const float* __restrict__ dis, const float* __restrict__ ew,
                          const float* __restrict__ al, const unsigned* __restrict__ mxe,
                          const float* __restrict__ den, const int* __restrict__ row,
                          const int* __restrict__ col, float* __restrict__ acc, int E)
{
  size_t idx = (size_t)blockIdx.x*blockDim.x + threadIdx.x;
  size_t tot = (size_t)E * HDIM;
  if (idx >= tot) return;
  int e = (int)(idx / HDIM);
  int c = (int)(idx % HDIM);
  int h = c / GHD;
  int r = row[e], cl = col[e];
  float wd  = dis[r]*ew[e]*dis[cl];
  float att = expf(al[(size_t)e*GATH+h] - fdec(mxe[(size_t)cl*GATH+h]))
              / (den[(size_t)cl*GATH+h] + 1e-16f);
  atomicAdd(&acc[(size_t)cl*HDIM+c], xw[(size_t)r*HDIM+c]*wd + gg[(size_t)r*HDIM+c]*att);
}

// ---------------- pyramid pooling ----------------
__global__ void k_pool_w(float* w)
{
  int lev = blockIdx.x;
  const int levels[4] = {1,2,4,8};
  int outn = SEQL / levels[lev];
  for (int j = threadIdx.x; j < SEQL; j += blockDim.x){
    float acc = 0.f;
    for (int i=0;i<outn;++i){
      int s = (i*SEQL)/outn;
      int e = ((i+1)*SEQL + outn - 1)/outn;
      if (j >= s && j < e) acc += 1.0f/(float)(e - s);
    }
    w[lev*SEQL + j] = acc/(float)outn;
  }
}

__global__ void k_pool_apply(const float* __restrict__ X, const float* __restrict__ w,
                             float* __restrict__ pooled)
{
  int idx = blockIdx.x*blockDim.x + threadIdx.x;
  if (idx >= 4*NB*HDIM) return;
  int lev = idx / (NB*HDIM);
  int r = idx - lev*NB*HDIM;
  int b = r / HDIM, c = r % HDIM;
  float s = 0.f;
  for (int l=0;l<SEQL;++l) s += w[lev*SEQL+l] * X[((size_t)b*SEQL+l)*HDIM + c];
  pooled[idx] = s;
}

// ---------------- attention (L=203, hd=28) ----------------
__global__ __launch_bounds__(256)
void k_attn(const float* __restrict__ qkv, float* __restrict__ ao)
{
  int bh = blockIdx.x;
  int b = bh >> 4, h = bh & 15;
  int i = threadIdx.x;
  if (i >= SEQL) return;
  const float* qr = qkv + ((size_t)(b*SEQL + i))*D3 + h*HD;
  float q[HD];
  #pragma unroll
  for (int d=0; d<HD; ++d) q[d] = qr[d];
  const float scale = 0.18898223650461363f;  // 1/sqrt(28)
  float mx = -1e30f;
  for (int j=0; j<SEQL; ++j){
    const float* kr = qkv + ((size_t)(b*SEQL + j))*D3 + HDIM + h*HD;
    float s = 0.f;
    #pragma unroll
    for (int d=0; d<HD; ++d) s += q[d]*kr[d];
    mx = fmaxf(mx, s*scale);
  }
  float den = 0.f;
  float num[HD];
  #pragma unroll
  for (int d=0; d<HD; ++d) num[d] = 0.f;
  for (int j=0; j<SEQL; ++j){
    const float* kr = qkv + ((size_t)(b*SEQL + j))*D3 + HDIM + h*HD;
    float s = 0.f;
    #pragma unroll
    for (int d=0; d<HD; ++d) s += q[d]*kr[d];
    float p = expf(s*scale - mx);
    den += p;
    const float* vr = qkv + ((size_t)(b*SEQL + j))*D3 + 2*HDIM + h*HD;
    #pragma unroll
    for (int d=0; d<HD; ++d) num[d] += p*vr[d];
  }
  float* o = ao + ((size_t)(b*SEQL + i))*HDIM + h*HD;
  float inv = 1.f/den;
  for (int d=0; d<HD; ++d) o[d] = num[d]*inv;
}

// ---------------- gate softmax + fuse ----------------
__global__ void k_gate_fuse(const float* __restrict__ gf, const float* __restrict__ W2,
                            const float* __restrict__ b2, const float* __restrict__ gi,
                            float* __restrict__ fused)
{
  int b = threadIdx.x;
  if (b >= NB) return;
  float l0 = b2[0], l1 = b2[1];
  for (int k=0;k<HDIM;++k){
    float v = gf[(size_t)b*HDIM + k];
    l0 += v * W2[k*2]; l1 += v * W2[k*2+1];
  }
  float m = fmaxf(l0,l1);
  float e0 = expf(l0-m), e1 = expf(l1-m);
  float inv = 1.f/(e0+e1);
  float g0 = e0*inv, g1 = e1*inv;
  for (int c=0;c<HDIM;++c){
    fused[(size_t)b*896 + c]       = gi[(size_t)b*896 + c] * g0;
    fused[(size_t)b*896 + 448 + c] = gi[(size_t)b*896 + 448 + c] * g1;
  }
}

__global__ void k_final(const float* __restrict__ f, const float* __restrict__ W,
                        const float* __restrict__ bias, float* __restrict__ out)
{
  int b = threadIdx.x; if (b >= NB) return;
  float s = bias[0];
  for (int k=0;k<64;++k) s += f[(size_t)b*64+k]*W[k];
  out[b] = sigm(s);
}

// =================== host ===================
extern "C" void kernel_launch(void* const* d_in, const int* in_sizes, int n_in,
                              void* d_out, int out_size, void* d_ws, size_t ws_size,
                              hipStream_t stream)
{
  (void)n_in; (void)out_size; (void)ws_size;
  const float* data = (const float*)d_in[0];
  const int*   eidx = (const int*)d_in[1];
  const int E = in_sizes[1] / 2;
  const int* rowp = eidx;
  const int* colp = eidx + E;
  auto F = [&](int i)->const float*{ return (const float*)d_in[i]; };

  // ---- workspace (bump allocator) ----
  char* base = (char*)d_ws; size_t off = 0;
  auto alloc = [&](size_t elems)->void*{
    void* p = base + off; off += (elems*4 + 255) & ~(size_t)255; return p;
  };
  float*    xp    = (float*)alloc((size_t)NB*PSEQ*INDIM);
  unsigned* Bh    = (unsigned*)alloc(704*1344);          // packed weight scratch
  float*    xg    = (float*)alloc((size_t)NNODE*HDIM);
  float*    hb    = (float*)alloc((size_t)NNODE*HDIM);
  float*    xw    = (float*)alloc((size_t)NNODE*HDIM);
  float*    gg    = (float*)alloc((size_t)NNODE*HDIM);
  float*    accb  = (float*)alloc((size_t)NNODE*HDIM);
  float*    hsck  = (float*)alloc((size_t)NNODE*HDIM);
  float*    tfh   = (float*)alloc((size_t)NNODE*HDIM);
  float*    aob   = (float*)alloc((size_t)NNODE*HDIM);
  float*    tbig  = (float*)alloc((size_t)NNODE*1792);
  float*    eA    = (float*)alloc((size_t)NNODE*64);
  float*    eB    = (float*)alloc((size_t)NNODE*64);
  float*    ew    = (float*)alloc((size_t)E);
  float*    deg   = (float*)alloc((size_t)NNODE);
  float*    a_s   = (float*)alloc((size_t)NNODE*GATH);
  float*    a_d   = (float*)alloc((size_t)NNODE*GATH);
  float*    als   = (float*)alloc((size_t)NNODE*GATH);
  float*    den   = (float*)alloc((size_t)NNODE*GATH);
  unsigned* mxe   = (unsigned*)alloc((size_t)NNODE*GATH);
  float*    al    = (float*)alloc((size_t)E*GATH);
  float*    pooled= (float*)alloc((size_t)4*NB*HDIM);
  float*    poolw = (float*)alloc((size_t)4*SEQL);
  float*    cat   = (float*)alloc((size_t)NB*D3);
  float*    gnnf  = (float*)alloc((size_t)NB*HDIM);
  float*    sckf  = (float*)alloc((size_t)NB*HDIM);
  float*    sea   = (float*)alloc((size_t)NB*HDIM);
  float*    seh   = (float*)alloc((size_t)NB*28);
  float*    ses   = (float*)alloc((size_t)NB*HDIM);
  float*    gi    = (float*)alloc((size_t)NB*896);
  float*    gf    = (float*)alloc((size_t)NB*HDIM);
  float*    fusedb= (float*)alloc((size_t)NB*896);
  float*    sA    = (float*)alloc((size_t)NB*D3);
  float*    sB    = (float*)alloc((size_t)NB*D3);

  auto packw = [&](const float* W, int K, int n, int ldw, int col0){
    int tot = (K/2)*n;
    k_pack_w<<<dim3((tot+255)/256),256,0,stream>>>(W, Bh, K, n, ldw, col0);
  };
  auto gemm = [&](const float* A, const float* bias, const float* res,
                  const float* bng_, const float* bnb_, float* C,
                  int M,int N,int K,int lda,int ldc,int ccol0,int accin,int epi,
                  int aseq,int apseq,int aoff){
    dim3 g((N+63)/64, (M+63)/64);
    k_gemm<<<g,128,0,stream>>>(A, Bh, bias, res, bng_, bnb_, C,
                               M,N,K,lda,ldc,ccol0,accin,epi,aseq,apseq,aoff);
  };

  // ---- shared preprocessing ----
  hipMemsetAsync(xp, 0, (size_t)NB*PSEQ*INDIM*4, stream);
  k_build_xp<<<((NB*SEQL*INDIM)+255)/256,256,0,stream>>>(data, xp);
  k_pool_w<<<4,256,0,stream>>>(poolw);

  // =============== GNN backbone ===============
  packw(F(48), INDIM, HDIM, HDIM, 0);                         // in_proj.W
  gemm(xp, F(49), nullptr,nullptr,nullptr, xg, NNODE,HDIM,INDIM, INDIM,HDIM,0, 0,1, SEQL,PSEQ,7);
  k_ln<<<NNODE,256,0,stream>>>(xg, F(47), F(46), xg, HDIM, 1); // in_ln + gelu

  for (int li=0; li<6; ++li){
    int pl = 50 + 20*li;
    k_ln<<<NNODE,256,0,stream>>>(xg, F(pl+17), F(pl+16), hb, HDIM, 0);  // pre
    // edge weight MLP (split concat GEMM)
    packw(F(pl+0), HDIM, 64, 64, 0);
    gemm(hb, F(pl+1), nullptr,nullptr,nullptr, eA, NNODE,64,HDIM, HDIM,64,0, 0,1, 0,0,0);
    packw(F(pl+0) + (size_t)HDIM*64, HDIM, 64, 64, 0);
    gemm(hb, nullptr, nullptr,nullptr,nullptr, eB, NNODE,64,HDIM, HDIM,64,0, 0,0, 0,0,0);
    k_edge_mlp<<<(E+255)/256,256,0,stream>>>(eA,eB,rowp,colp,F(pl+2),F(pl+3),ew,E);
    // degree / norm
    k_fill<<<(NNODE+255)/256,256,0,stream>>>(deg, 1.0f, NNODE);
    k_deg_edge<<<(E+255)/256,256,0,stream>>>(deg, colp, ew, E);
    k_deg_fin<<<(NNODE+255)/256,256,0,stream>>>(deg, NNODE);   // deg -> dis
    // GCN / GAT projections
    packw(F(pl+12), HDIM, HDIM, HDIM, 0);
    gemm(hb, nullptr,nullptr,nullptr,nullptr, xw, NNODE,HDIM,HDIM, HDIM,HDIM,0, 0,0, 0,0,0);
    packw(F(pl+8), HDIM, HDIM, HDIM, 0);
    gemm(hb, nullptr,nullptr,nullptr,nullptr, gg, NNODE,HDIM,HDIM, HDIM,HDIM,0, 0,0, 0,0,0);
    // GAT attention
    k_asad<<<((NNODE*GATH)+255)/256,256,0,stream>>>(gg, F(pl+10), F(pl+9), a_s, a_d, NNODE);
    k_gat_self<<<((NNODE*GATH)+255)/256,256,0,stream>>>(a_s, a_d, als, mxe, NNODE);
    k_gat_edge<<<(E+255)/256,256,0,stream>>>(a_s, a_d, rowp, colp, al, mxe, E);
    k_den_init<<<((NNODE*GATH)+255)/256,256,0,stream>>>(als, mxe, den, NNODE);
    k_den_edge<<<((E*GATH)+255)/256,256,0,stream>>>(al, colp, mxe, den, E);
    // combined self-term init + edge scatter (GCN + GAT)
    k_acc_init<<<((NNODE*HDIM)+255)/256,256,0,stream>>>(xw, gg, deg, als, mxe, den,
                                                        F(pl+13), F(pl+11), accb, NNODE);
    {
      size_t tot = (size_t)E*HDIM;
      k_scatter<<<(unsigned)((tot+255)/256),256,0,stream>>>(xw, gg, deg, ew, al, mxe, den,
                                                            rowp, colp, accb, E);
    }
    k_addip<<<((NNODE*HDIM)+255)/256,256,0,stream>>>(xg, accb, NNODE*HDIM);
    // FFN + residual
    k_ln<<<NNODE,256,0,stream>>>(xg, F(pl+15), F(pl+14), hb, HDIM, 0);  // post
    packw(F(pl+4), HDIM, 1792, 1792, 0);
    gemm(hb, F(pl+5), nullptr,nullptr,nullptr, tbig, NNODE,1792,HDIM, HDIM,1792,0, 0,2, 0,0,0);
    packw(F(pl+6), 1792, HDIM, HDIM, 0);
    gemm(tbig, F(pl+7), xg, nullptr,nullptr, xg, NNODE,HDIM,1792, 1792,HDIM,0, 0,3, 0,0,0);
    // SE
    k_colmean<<<((NB*HDIM)+255)/256,256,0,stream>>>(xg, sea, NB, SEQL, HDIM, HDIM, 0);
    k_small_nn<<<((NB*28)+255)/256,256,0,stream>>>(sea, F(pl+18), nullptr, seh, NB,28,HDIM, 1);
    k_small_nn<<<((NB*HDIM)+255)/256,256,0,stream>>>(seh, F(pl+19), nullptr, ses, NB,HDIM,28, 2);
    k_scale_rows<<<((NNODE*HDIM)+255)/256,256,0,stream>>>(xg, xg, ses, SEQL, HDIM, NNODE*HDIM);
  }

  // =============== SCK backbone ===============
  // conv bank as 63 accumulated tap-GEMMs over padded input
  const int KSarr[7] = {3,5,7,9,11,13,15};
  for (int ci=0; ci<7; ++ci){
    int ks = KSarr[ci];
    const float* Wc  = F(182+4*ci);
    const float* bc  = F(182+4*ci+1);
    const float* bnb_= F(182+4*ci+2);
    const float* bng_= F(182+4*ci+3);
    for (int t=0; t<ks; ++t){
      k_pack_conv<<<(((INDIM/2)*64)+255)/256,256,0,stream>>>(Wc, Bh, ks, t);
      gemm(xp, (t==ks-1)?bc:nullptr, nullptr, bng_, bnb_, hsck,
           NNODE, 64, INDIM, INDIM, HDIM, ci*64, (t>0)?1:0, (t==ks-1)?4:0,
           SEQL, PSEQ, 7 + t - ks/2);
    }
  }
  // SE on conv bank
  k_colmean<<<((NB*HDIM)+255)/256,256,0,stream>>>(hsck, sea, NB, SEQL, HDIM, HDIM, 0);
  k_small_nn<<<((NB*28)+255)/256,256,0,stream>>>(sea, F(222), nullptr, seh, NB,28,HDIM, 1);
  k_small_nn<<<((NB*HDIM)+255)/256,256,0,stream>>>(seh, F(223), nullptr, ses, NB,HDIM,28, 2);
  k_scale_rows<<<((NNODE*HDIM)+255)/256,256,0,stream>>>(tfh, hsck, ses, SEQL, HDIM, NNODE*HDIM);
  // 4 transformer layers
  for (int tl=0; tl<4; ++tl){
    int ti = 224 + 12*tl;
    k_ln<<<NNODE,256,0,stream>>>(tfh, F(ti+5), F(ti+4), hb, HDIM, 0);
    packw(F(ti+10), HDIM, D3, D3, 0);
    gemm(hb, F(ti+11), nullptr,nullptr,nullptr, tbig, NNODE,D3,HDIM, HDIM,D3,0, 0,1, 0,0,0);
    k_attn<<<NB*NHEADS,256,0,stream>>>(tbig, aob);
    packw(F(ti+8), HDIM, HDIM, HDIM, 0);
    gemm(aob, F(ti+9), tfh, nullptr,nullptr, tfh, NNODE,HDIM,HDIM, HDIM,HDIM,0, 0,3, 0,0,0);
    k_ln<<<NNODE,256,0,stream>>>(tfh, F(ti+7), F(ti+6), hb, HDIM, 0);
    packw(F(ti+0), HDIM, 1792, 1792, 0);
    gemm(hb, F(ti+1), nullptr,nullptr,nullptr, tbig, NNODE,1792,HDIM, HDIM,1792,0, 0,2, 0,0,0);
    packw(F(ti+2), 1792, HDIM, HDIM, 0);
    gemm(tbig, F(ti+3), tfh, nullptr,nullptr, tfh, NNODE,HDIM,1792, 1792,HDIM,0, 0,3, 0,0,0);
  }

  // =============== pyramids ===============
  auto pyramid = [&](const float* xbuf, int fusWi, int fusbi, int lnbi, int lngi,
                     int projBase, float* feat){
    k_pool_apply<<<((4*NB*HDIM)+255)/256,256,0,stream>>>(xbuf, poolw, pooled);
    for (int lev=0; lev<4; ++lev){
      packw(F(projBase+2*lev), HDIM, 112, 112, 0);
      gemm(pooled + (size_t)lev*NB*HDIM, F(projBase+2*lev+1), nullptr,nullptr,nullptr,
           cat, NB, 112, HDIM, HDIM, D3, lev*112, 0, 1, 0,0,0);
    }
    k_colmean<<<((NB*HDIM)+255)/256,256,0,stream>>>(xbuf, cat, NB, SEQL, HDIM, D3, 448);
    k_colmax <<<((NB*HDIM)+255)/256,256,0,stream>>>(xbuf, cat, NB, SEQL, HDIM, D3, 896);
    packw(F(fusWi), D3, HDIM, HDIM, 0);
    gemm(cat, F(fusbi), nullptr,nullptr,nullptr, feat, NB,HDIM,D3, D3,HDIM,0, 0,1, 0,0,0);
    k_ln<<<NB,256,0,stream>>>(feat, F(lngi), F(lnbi), feat, HDIM, 1);
  };
  pyramid(xg,  170,171, 172,173, 174, gnnf);
  pyramid(tfh, 210,211, 212,213, 214, sckf);

  // =============== cross attention (1-token softmax == identity) ===============
  // ge = (sckf @ Wv_ca_gnn + bv) @ Wout + bout -> gi[:, :448]
  packw(F(4), HDIM, HDIM, D3, 896);
  gemm(sckf, F(5)+896, nullptr,nullptr,nullptr, sA, NB,HDIM,HDIM, HDIM,HDIM,0, 0,1, 0,0,0);
  packw(F(2), HDIM, HDIM, HDIM, 0);
  gemm(sA, F(3), nullptr,nullptr,nullptr, gi, NB,HDIM,HDIM, HDIM,896,0, 0,1, 0,0,0);
  // se_ = (gnnf @ Wv_ca_sck + bv) @ Wout + bout -> gi[:, 448:]
  packw(F(8), HDIM, HDIM, D3, 896);
  gemm(gnnf, F(9)+896, nullptr,nullptr,nullptr, sA, NB,HDIM,HDIM, HDIM,HDIM,0, 0,1, 0,0,0);
  packw(F(6), HDIM, HDIM, HDIM, 0);
  gemm(sA, F(7), nullptr,nullptr,nullptr, gi, NB,HDIM,HDIM, HDIM,896,448, 0,1, 0,0,0);

  // gate
  packw(F(40), 896, HDIM, HDIM, 0);
  gemm(gi, F(41), nullptr,nullptr,nullptr, gf, NB,HDIM,896, 896,HDIM,0, 0,1, 0,0,0);
  k_ln<<<NB,256,0,stream>>>(gf, F(45), F(44), gf, HDIM, 1);
  k_gate_fuse<<<1,64,0,stream>>>(gf, F(42), F(43), gi, fusedb);

  // fusion MLP
  packw(F(28), 896, D3, D3, 0);
  gemm(fusedb, F(29), nullptr,nullptr,nullptr, sA, NB,D3,896, 896,D3,0, 0,1, 0,0,0);
  k_ln<<<NB,256,0,stream>>>(sA, F(35), F(34), sA, D3, 1);
  packw(F(30), D3, 896, 896, 0);
  gemm(sA, F(31), nullptr,nullptr,nullptr, sB, NB,896,D3, D3,896,0, 0,1, 0,0,0);
  k_ln<<<NB,256,0,stream>>>(sB, F(37), F(36), sB, 896, 1);
  packw(F(32), 896, HDIM, HDIM, 0);
  gemm(sB, F(33), nullptr,nullptr,nullptr, sA, NB,HDIM,896, 896,HDIM,0, 0,1, 0,0,0);
  k_ln<<<NB,256,0,stream>>>(sA, F(39), F(38), sA, HDIM, 1);

  // classifier
  packw(F(10), HDIM, 512, 512, 0);
  gemm(sA, F(11), nullptr,nullptr,nullptr, sB, NB,512,HDIM, HDIM,512,0, 0,1, 0,0,0);
  k_ln<<<NB,256,0,stream>>>(sB, F(21), F(20), sB, 512, 1);
  packw(F(12), 512, 256, 256, 0);
  gemm(sB, F(13), nullptr,nullptr,nullptr, sA, NB,256,512, 512,256,0, 0,1, 0,0,0);
  k_ln<<<NB,256,0,stream>>>(sA, F(23), F(22), sA, 256, 1);
  packw(F(14), 256, 128, 128, 0);
  gemm(sA, F(15), nullptr,nullptr,nullptr, sB, NB,128,256, 256,128,0, 0,1, 0,0,0);
  k_ln<<<NB,256,0,stream>>>(sB, F(25), F(24), sB, 128, 1);
  packw(F(16), 128, 64, 64, 0);
  gemm(sB, F(17), nullptr,nullptr,nullptr, sA, NB,64,128, 128,64,0, 0,1, 0,0,0);
  k_ln<<<NB,256,0,stream>>>(sA, F(27), F(26), sA, 64, 1);
  k_final<<<1,64,0,stream>>>(sA, F(18), F(19), (float*)d_out);
}
